// MegaBlockGroupedFeedForward_57097295233530
// MI455X (gfx1250) — matrix operand order repaired and checked by reading, not compiled, not run on hardware
//
#include <hip/hip_runtime.h>
#include <stdint.h>

// ---------------------------------------------------------------------------
// Megablocks grouped MoE FFN (SwiGLU) for gfx1250 (MI455X), f16 WMMA path.
//   Phase 1: gated[e] = silu(x_e @ w1_eT) * (x_e @ w3_eT)   -> f16 ws
//   Phase 2: out[e]   = gated[e] @ w2_e                      -> f32 d_out
// Double-buffered LDS, register-staged global loads, async A-tile copies.
// Block order puts mt fastest so weight tiles are L2-resident across blocks.
// ---------------------------------------------------------------------------

typedef __attribute__((ext_vector_type(16))) _Float16 v16h;
typedef __attribute__((ext_vector_type(2)))  __fp16   v2fp16;   // cvt_pkrtz return type
typedef __attribute__((ext_vector_type(8)))  float    v8f;

#define E_ 8
#define T_ 1024
#define D_ 2048
#define H_ 4096

#define BM 128
#define BN 128
#define BK 32
#define NTHREADS 512          // 16 waves (wave32)
#define LDSS 40               // 32 + 8 pad halves -> 80B row stride (16B aligned)

union Frag16 { v16h v; uint4 u[2]; };

// Pack two f32 -> packed f16 pair in a single v_cvt_pk_rtz_f16_f32.
__device__ __forceinline__ unsigned pack2h(float lo, float hi) {
    union { v2fp16 v; unsigned u; } c;
    c.v = __builtin_amdgcn_cvt_pkrtz(lo, hi);
    return c.u;
}

__device__ __forceinline__ unsigned short f2h_bits(float f) {
    _Float16 h = (_Float16)f;
    union { _Float16 h; unsigned short u; } c; c.h = h; return c.u;
}

__device__ __forceinline__ float silu_f(float a) {
    return a / (1.0f + __expf(-a));
}

// A fragment: 16x32 f16. lane&15 -> M row; lane>>4 selects K groups
// {0..7,16..23} vs {8..15,24..31} per the CDNA5 A-matrix striping.
__device__ __forceinline__ v16h load_frag_a(const unsigned short* lds, int mbase, int lane) {
    const int row = mbase + (lane & 15);
    const int kg  = (lane >> 4) << 3;                         // 0 or 8
    const unsigned short* p = lds + row * LDSS + kg;
    Frag16 f;
    f.u[0] = *reinterpret_cast<const uint4*>(p);              // K = kg .. kg+7
    f.u[1] = *reinterpret_cast<const uint4*>(p + 16);         // K = kg+16 .. kg+23
    return f.v;
}

// B fragment: 32x16 f16 staged in LDS as [N][K]. lane&15 -> N col;
// lane>>4 selects K groups {0..7,16..23} vs {8..15,24..31}.
__device__ __forceinline__ v16h load_frag_b(const unsigned short* lds, int nbase, int lane) {
    const int col = nbase + (lane & 15);
    const int kg  = (lane >> 4) << 3;                         // 0 or 8
    const unsigned short* p = lds + col * LDSS + kg;
    Frag16 f;
    f.u[0] = *reinterpret_cast<const uint4*>(p);
    f.u[1] = *reinterpret_cast<const uint4*>(p + 16);
    return f.v;
}

__device__ __forceinline__ v8f wmma_f16(v16h a, v16h b, v8f c) {
    return __builtin_amdgcn_wmma_f32_16x16x32_f16(false, a, false, b, (short)0, c, false, false);
}

// Async global -> LDS 128-bit copy (CDNA5 async path, tracked by ASYNCcnt).
__device__ __forceinline__ void async_copy_b128(unsigned lds_off, const void* gaddr) {
    asm volatile("global_load_async_to_lds_b128 %0, %1, off"
                 :: "v"(lds_off), "v"(gaddr) : "memory");
}
__device__ __forceinline__ void wait_async0() {
    asm volatile("s_wait_asynccnt 0x0" ::: "memory");
}

// ---------------------------------------------------------------------------
// Phase 1: h1 = x@w1^T, h3 = x@w3^T (NT), fused SwiGLU -> f16 gated ws
// ---------------------------------------------------------------------------
__global__ __launch_bounds__(NTHREADS)
void moe_swiglu_phase1(const float* __restrict__ x,
                       const float* __restrict__ w1,
                       const float* __restrict__ w3,
                       unsigned short* __restrict__ gated) {
    __shared__ unsigned short sA [2][BM * LDSS];
    __shared__ unsigned short sB1[2][BN * LDSS];
    __shared__ unsigned short sB3[2][BN * LDSS];

    const int MT = T_ / BM;                   // 8
    const int NT = H_ / BN;                   // 32
    const int bid = blockIdx.x;
    const int e   = bid / (MT * NT);
    const int rem = bid % (MT * NT);
    const int nt  = rem / MT;                 // slow: weight tile index
    const int mt  = rem % MT;                 // fast: adjacent blocks reuse weights in L2

    const float* xA  = x  + (size_t)(e * T_ + mt * BM) * D_;
    const float* pW1 = w1 + (size_t)(e * H_ + nt * BN) * D_;
    const float* pW3 = w3 + (size_t)(e * H_ + nt * BN) * D_;

    const int tid  = threadIdx.x;
    const int lane = tid & 31;
    const int wave = tid >> 5;
    const int wr   = (wave >> 2) * 32;
    const int wc   = (wave & 3) * 32;

    // staging coordinates (2 float4 slots per thread per tile)
    int srow[2], sc4[2];
    #pragma unroll
    for (int it = 0; it < 2; ++it) {
        const int slot = tid + it * NTHREADS;
        srow[it] = slot >> 3;                 // 8 float4 per 32-wide row
        sc4[it]  = (slot & 7) << 2;
    }

    float4 rA[2], rB1[2], rB3[2];

    auto stage_load = [&](int k0) {
        #pragma unroll
        for (int it = 0; it < 2; ++it) {
            const size_t goff = (size_t)srow[it] * D_ + k0 + sc4[it];
            rA[it]  = *reinterpret_cast<const float4*>(xA  + goff);
            rB1[it] = *reinterpret_cast<const float4*>(pW1 + goff);
            rB3[it] = *reinterpret_cast<const float4*>(pW3 + goff);
        }
    };
    auto stage_store = [&](int p) {
        #pragma unroll
        for (int it = 0; it < 2; ++it) {
            const int off = srow[it] * LDSS + sc4[it];
            uint2 a  = { pack2h(rA[it].x,  rA[it].y),  pack2h(rA[it].z,  rA[it].w)  };
            uint2 b1 = { pack2h(rB1[it].x, rB1[it].y), pack2h(rB1[it].z, rB1[it].w) };
            uint2 b3 = { pack2h(rB3[it].x, rB3[it].y), pack2h(rB3[it].z, rB3[it].w) };
            *reinterpret_cast<uint2*>(&sA [p][off]) = a;
            *reinterpret_cast<uint2*>(&sB1[p][off]) = b1;
            *reinterpret_cast<uint2*>(&sB3[p][off]) = b3;
        }
    };

    v8f acc1[2][2], acc3[2][2];
    #pragma unroll
    for (int i = 0; i < 2; ++i)
        #pragma unroll
        for (int j = 0; j < 2; ++j) {
            acc1[i][j] = (v8f){0.f,0.f,0.f,0.f,0.f,0.f,0.f,0.f};
            acc3[i][j] = (v8f){0.f,0.f,0.f,0.f,0.f,0.f,0.f,0.f};
        }

    stage_load(0);
    stage_store(0);
    __syncthreads();

    const int NSTEPS = D_ / BK;               // 64
    for (int s = 0; s < NSTEPS; ++s) {
        const int  p    = s & 1;
        const bool more = (s + 1) < NSTEPS;
        if (more) stage_load((s + 1) * BK);   // prefetch next tiles into regs

        const v16h a0  = load_frag_a(sA [p], wr +  0, lane);
        const v16h a1  = load_frag_a(sA [p], wr + 16, lane);
        const v16h b10 = load_frag_b(sB1[p], wc +  0, lane);
        const v16h b11 = load_frag_b(sB1[p], wc + 16, lane);
        const v16h b30 = load_frag_b(sB3[p], wc +  0, lane);
        const v16h b31 = load_frag_b(sB3[p], wc + 16, lane);

        if (more) stage_store(p ^ 1);         // fill other buffer

        acc1[0][0] = wmma_f16(a0, b10, acc1[0][0]);
        acc1[0][1] = wmma_f16(a0, b11, acc1[0][1]);
        acc1[1][0] = wmma_f16(a1, b10, acc1[1][0]);
        acc1[1][1] = wmma_f16(a1, b11, acc1[1][1]);
        acc3[0][0] = wmma_f16(a0, b30, acc3[0][0]);
        acc3[0][1] = wmma_f16(a0, b31, acc3[0][1]);
        acc3[1][0] = wmma_f16(a1, b30, acc3[1][0]);
        acc3[1][1] = wmma_f16(a1, b31, acc3[1][1]);
        __syncthreads();
    }

    // fused SwiGLU epilogue, f16 store to gated workspace
    const int mrow0 = mt * BM + wr + ((lane >> 4) << 3);
    const int ncol  = nt * BN + wc + (lane & 15);
    #pragma unroll
    for (int i = 0; i < 2; ++i)
        #pragma unroll
        for (int j = 0; j < 2; ++j)
            #pragma unroll
            for (int r = 0; r < 8; ++r) {
                const float g = silu_f(acc1[i][j][r]) * acc3[i][j][r];
                const int m = mrow0 + i * 16 + r;
                const int n = ncol  + j * 16;
                gated[(size_t)(e * T_ + m) * H_ + n] = f2h_bits(g);
            }
}

// ---------------------------------------------------------------------------
// Phase 2: out = gated @ w2 (NN).  A tile via async global->LDS copies;
// w2 tile transposed fp32->f16 into LDS ([N][K]).
// ---------------------------------------------------------------------------
__global__ __launch_bounds__(NTHREADS)
void moe_swiglu_phase2(const unsigned short* __restrict__ gated,
                       const float* __restrict__ w2,
                       float* __restrict__ out) {
    __shared__ unsigned short sA[2][BM * LDSS];
    __shared__ unsigned short sB[2][BN * LDSS];

    const int MT = T_ / BM;                   // 8
    const int NT = D_ / BN;                   // 16
    const int bid = blockIdx.x;
    const int e   = bid / (MT * NT);
    const int rem = bid % (MT * NT);
    const int nt  = rem / MT;                 // slow: w2 tile index
    const int mt  = rem % MT;                 // fast: reuse w2 tile in L2

    const unsigned short* gA = gated + (size_t)(e * T_ + mt * BM) * H_;
    const float*          pW = w2    + (size_t)(e * H_) * D_ + nt * BN;

    const int tid  = threadIdx.x;
    const int lane = tid & 31;
    const int wave = tid >> 5;
    const int wr   = (wave >> 2) * 32;
    const int wc   = (wave & 3) * 32;

    // A tile: one 16B async copy per thread (512 x 16B = 128x32 f16)
    const int arow = tid >> 2;
    const int ac8  = (tid & 3) << 3;
    auto a_async = [&](int k0, int p) {
        const unsigned short* g = gA + (size_t)arow * H_ + k0 + ac8;
        const unsigned lds = (unsigned)(unsigned long long)&sA[p][arow * LDSS + ac8];
        async_copy_b128(lds, g);
    };

    // B tile: fp32 [K][N] in memory -> f16 [N][K] in LDS (transpose on store)
    int bkr[2], bc4[2];
    #pragma unroll
    for (int it = 0; it < 2; ++it) {
        const int slot = tid + it * NTHREADS;
        bkr[it] = slot >> 5;                  // 32 float4 per 128-wide row
        bc4[it] = (slot & 31) << 2;
    }
    float4 rB[2];
    auto b_load = [&](int k0) {
        #pragma unroll
        for (int it = 0; it < 2; ++it)
            rB[it] = *reinterpret_cast<const float4*>(pW + (size_t)(k0 + bkr[it]) * D_ + bc4[it]);
    };
    auto b_store = [&](int p) {
        #pragma unroll
        for (int it = 0; it < 2; ++it) {
            sB[p][(bc4[it] + 0) * LDSS + bkr[it]] = f2h_bits(rB[it].x);
            sB[p][(bc4[it] + 1) * LDSS + bkr[it]] = f2h_bits(rB[it].y);
            sB[p][(bc4[it] + 2) * LDSS + bkr[it]] = f2h_bits(rB[it].z);
            sB[p][(bc4[it] + 3) * LDSS + bkr[it]] = f2h_bits(rB[it].w);
        }
    };

    v8f acc[2][2];
    #pragma unroll
    for (int i = 0; i < 2; ++i)
        #pragma unroll
        for (int j = 0; j < 2; ++j)
            acc[i][j] = (v8f){0.f,0.f,0.f,0.f,0.f,0.f,0.f,0.f};

    a_async(0, 0);
    b_load(0);
    b_store(0);
    wait_async0();
    __syncthreads();

    const int NSTEPS = H_ / BK;               // 128
    for (int s = 0; s < NSTEPS; ++s) {
        const int  p    = s & 1;
        const bool more = (s + 1) < NSTEPS;
        if (more) {
            a_async((s + 1) * BK, p ^ 1);     // async fill of other A buffer
            b_load((s + 1) * BK);
        }

        const v16h a0 = load_frag_a(sA[p], wr +  0, lane);
        const v16h a1 = load_frag_a(sA[p], wr + 16, lane);
        const v16h b0 = load_frag_b(sB[p], wc +  0, lane);
        const v16h b1 = load_frag_b(sB[p], wc + 16, lane);

        if (more) b_store(p ^ 1);

        acc[0][0] = wmma_f16(a0, b0, acc[0][0]);
        acc[0][1] = wmma_f16(a0, b1, acc[0][1]);
        acc[1][0] = wmma_f16(a1, b0, acc[1][0]);
        acc[1][1] = wmma_f16(a1, b1, acc[1][1]);

        wait_async0();                        // A writes to p^1 landed
        __syncthreads();
    }

    const int mrow0 = mt * BM + wr + ((lane >> 4) << 3);
    const int ncol  = nt * BN + wc + (lane & 15);
    #pragma unroll
    for (int i = 0; i < 2; ++i)
        #pragma unroll
        for (int j = 0; j < 2; ++j)
            #pragma unroll
            for (int r = 0; r < 8; ++r) {
                const int m = mrow0 + i * 16 + r;
                const int n = ncol  + j * 16;
                out[(size_t)(e * T_ + m) * D_ + n] = acc[i][j][r];
            }
}

// ---------------------------------------------------------------------------
extern "C" void kernel_launch(void* const* d_in, const int* in_sizes, int n_in,
                              void* d_out, int out_size, void* d_ws, size_t ws_size,
                              hipStream_t stream) {
    const float* x  = (const float*)d_in[0];
    const float* w1 = (const float*)d_in[1];
    const float* w2 = (const float*)d_in[2];
    const float* w3 = (const float*)d_in[3];
    // d_in[4] = num_experts scalar; shapes are compile-time constants here.

    unsigned short* gated = (unsigned short*)d_ws;   // E*T*H f16 = 64 MiB
    float* out = (float*)d_out;

    dim3 blk(NTHREADS);
    dim3 g1(E_ * (T_ / BM) * (H_ / BN));   // 2048 blocks
    moe_swiglu_phase1<<<g1, blk, 0, stream>>>(x, w1, w3, gated);

    dim3 g2(E_ * (T_ / BM) * (D_ / BN));   // 1024 blocks
    moe_swiglu_phase2<<<g2, blk, 0, stream>>>(gated, w2, out);
}
